// ROIMaskHead_25589415149603
// MI455X (gfx1250) — matrix operand order repaired and checked by reading, not compiled, not run on hardware
//
#include <hip/hip_runtime.h>

typedef __bf16 bf16;
typedef __attribute__((ext_vector_type(8)))  __bf16 v8bf;
typedef __attribute__((ext_vector_type(16))) __bf16 v16bf;
typedef __attribute__((ext_vector_type(8)))  float  v8f;

union V16 { v16bf v; v8bf h[2]; };

// Padded activation layout: [roi][16][16][256] bf16, image at (1..14,1..14),
// zero halo. Per-ROI stride = 65536 elements.
#define ROISTRIDE 65536

// ---------------------------------------------------------------------------
// Bulk zero (clears both ping-pong buffers incl. halos), 16B per thread
// ---------------------------------------------------------------------------
__global__ __launch_bounds__(256) void zero_kernel(uint4* __restrict__ p, long n16) {
  long i = (long)blockIdx.x * 256 + threadIdx.x;
  if (i < n16) p[i] = uint4{0u, 0u, 0u, 0u};
}

// f32 NHWC -> bf16 padded layout, 4 channels per thread
__global__ __launch_bounds__(256) void cvt_pad_kernel(
    const float* __restrict__ in, bf16* __restrict__ out, int nroi) {
  long i = (long)blockIdx.x * 256 + threadIdx.x;
  long total = (long)nroi * 196 * 64;
  if (i >= total) return;
  int c4  = (int)(i & 63);
  int m   = (int)((i >> 6) % 196);
  int roi = (int)(i / (196 * 64));
  float4 f = ((const float4*)in)[i];
  struct B4 { bf16 a, b, c, d; } o;
  o.a = (bf16)f.x; o.b = (bf16)f.y; o.c = (bf16)f.z; o.d = (bf16)f.w;
  int oy = m / 14, ox = m % 14;
  ((B4*)(out + (long)roi * ROISTRIDE + ((oy + 1) * 16 + (ox + 1)) * 256))[c4] = o;
}

// Wc [4][3][3][256][256] (l,ky,kx,ic,oc) -> wcp [l][oc][kpos*256+ic]
__global__ __launch_bounds__(256) void pack_wc_kernel(
    const float* __restrict__ in, bf16* __restrict__ out) {
  long t = (long)blockIdx.x * 256 + threadIdx.x;
  if (t >= 4L * 9 * 256 * 256) return;
  int oc   = (int)(t & 255);
  int ic   = (int)((t >> 8) & 255);
  int kpos = (int)((t >> 16) % 9);
  int l    = (int)(t / (9L * 65536));
  out[((long)l * 256 + oc) * 2304 + kpos * 256 + ic] = (bf16)in[t];
}

// Wt [2][2][256][256] (ky,kx,ic,oc) -> wtp [p][oc][ic], p = ky*2+kx
__global__ __launch_bounds__(256) void pack_wt_kernel(
    const float* __restrict__ in, bf16* __restrict__ out) {
  long t = (long)blockIdx.x * 256 + threadIdx.x;
  if (t >= 4L * 256 * 256) return;
  int oc = (int)(t & 255);
  int ic = (int)((t >> 8) & 255);
  int p  = (int)(t >> 16);
  out[((long)p * 256 + oc) * 256 + ic] = (bf16)in[t];
}

// Wm [1][1][256][80] (ic,oc) -> wmp [oc][ic]
__global__ __launch_bounds__(256) void pack_wm_kernel(
    const float* __restrict__ in, bf16* __restrict__ out) {
  long t = (long)blockIdx.x * 256 + threadIdx.x;
  if (t >= 256L * 80) return;
  int oc = (int)(t % 80);
  int ic = (int)(t / 80);
  out[(long)oc * 256 + ic] = (bf16)in[t];
}

// ---------------------------------------------------------------------------
// 3x3 conv (SAME) as implicit GEMM: M=196 spatial, N=256 outC, K=2304.
// One wave per 32x64 output tile: 2 A frags x 4 B frags -> 8 WMMA per
// K-chunk from 12 b128 loads (1.5 loads/WMMA). Padded input -> branchless.
// ---------------------------------------------------------------------------
__global__ __launch_bounds__(256) void conv3x3_kernel(
    const bf16* __restrict__ xin, const bf16* __restrict__ wp,
    const float* __restrict__ bias, bf16* __restrict__ xout, int nroi) {
  const int lane = threadIdx.x & 31;
  const int wid  = threadIdx.x >> 5;
  long t = (long)blockIdx.x * 8 + wid;          // global tile id
  const int n_blk = (int)(t & 3); t >>= 2;      // 4 n-blocks of 64 channels
  const int m_blk = (int)(t % 7);               // 7 m-blocks of 32 rows
  const int roi   = (int)(t / 7);
  if (roi >= nroi) return;

  const int row = lane & 15;
  const int hi  = lane >> 4;
  const int koffA = hi * 8;                     // ISA 16-bit A layout
  const int koffB = hi * 8;                     // same k order as A

  int m0 = m_blk * 32 + row;       if (m0 > 195) m0 = 195;  // clamped rows
  int m1 = m_blk * 32 + 16 + row;  if (m1 > 195) m1 = 195;  // are discarded
  const int oy0 = m0 / 14, ox0 = m0 % 14;
  const int oy1 = m1 / 14, ox1 = m1 % 14;

  const bf16* __restrict__ base = xin + (long)roi * ROISTRIDE;
  const bf16* __restrict__ pa0 = base + ((oy0 + 1) * 16 + (ox0 + 1)) * 256 + koffA;
  const bf16* __restrict__ pa1 = base + ((oy1 + 1) * 16 + (ox1 + 1)) * 256 + koffA;
  const bf16* __restrict__ pb  = wp + (long)(n_blk * 64 + row) * 2304 + koffB;

  v8f acc[2][4] = {};

#pragma unroll
  for (int kpos = 0; kpos < 9; ++kpos) {
    const int ky = kpos / 3, kx = kpos % 3;
    const int aoff = ((ky - 1) * 16 + (kx - 1)) * 256;   // constant per kpos
#pragma unroll
    for (int cc = 0; cc < 256; cc += 32) {
      V16 A0, A1, B0, B1, B2, B3;
      A0.h[0] = *(const v8bf*)(pa0 + aoff + cc);
      A0.h[1] = *(const v8bf*)(pa0 + aoff + cc + 16);
      A1.h[0] = *(const v8bf*)(pa1 + aoff + cc);
      A1.h[1] = *(const v8bf*)(pa1 + aoff + cc + 16);
      const int bo = kpos * 256 + cc;           // all imm offsets off pb
      B0.h[0] = *(const v8bf*)(pb + bo);
      B0.h[1] = *(const v8bf*)(pb + bo + 16);
      B1.h[0] = *(const v8bf*)(pb + 16 * 2304 + bo);
      B1.h[1] = *(const v8bf*)(pb + 16 * 2304 + bo + 16);
      B2.h[0] = *(const v8bf*)(pb + 32 * 2304 + bo);
      B2.h[1] = *(const v8bf*)(pb + 32 * 2304 + bo + 16);
      B3.h[0] = *(const v8bf*)(pb + 48 * 2304 + bo);
      B3.h[1] = *(const v8bf*)(pb + 48 * 2304 + bo + 16);
      acc[0][0] = __builtin_amdgcn_wmma_f32_16x16x32_bf16(false, A0.v, false, B0.v, (short)0, acc[0][0], false, false);
      acc[0][1] = __builtin_amdgcn_wmma_f32_16x16x32_bf16(false, A0.v, false, B1.v, (short)0, acc[0][1], false, false);
      acc[0][2] = __builtin_amdgcn_wmma_f32_16x16x32_bf16(false, A0.v, false, B2.v, (short)0, acc[0][2], false, false);
      acc[0][3] = __builtin_amdgcn_wmma_f32_16x16x32_bf16(false, A0.v, false, B3.v, (short)0, acc[0][3], false, false);
      acc[1][0] = __builtin_amdgcn_wmma_f32_16x16x32_bf16(false, A1.v, false, B0.v, (short)0, acc[1][0], false, false);
      acc[1][1] = __builtin_amdgcn_wmma_f32_16x16x32_bf16(false, A1.v, false, B1.v, (short)0, acc[1][1], false, false);
      acc[1][2] = __builtin_amdgcn_wmma_f32_16x16x32_bf16(false, A1.v, false, B2.v, (short)0, acc[1][2], false, false);
      acc[1][3] = __builtin_amdgcn_wmma_f32_16x16x32_bf16(false, A1.v, false, B3.v, (short)0, acc[1][3], false, false);
    }
  }

  float bs[4];
#pragma unroll
  for (int s = 0; s < 4; ++s) bs[s] = bias[n_blk * 64 + s * 16 + row];
  bf16* __restrict__ ob = xout + (long)roi * ROISTRIDE + n_blk * 64 + row;

#pragma unroll
  for (int mg = 0; mg < 2; ++mg) {
    const int mrb = m_blk * 32 + mg * 16 + hi * 8;
    int ry = mrb / 14, rx = mrb - ry * 14;
#pragma unroll
    for (int r = 0; r < 8; ++r) {
      if (mrb + r < 196) {
        const int o = ((ry + 1) * 16 + rx + 1) * 256;
#pragma unroll
        for (int s = 0; s < 4; ++s) {
          float v = acc[mg][s][r] + bs[s];
          v = v > 0.f ? v : 0.f;
          ob[o + s * 16] = (bf16)v;
        }
      }
      ++rx;
      if (rx == 14) { rx = 0; ++ry; }
    }
  }
}

// ---------------------------------------------------------------------------
// Fused 2x2/stride-2 transpose conv (+bias+ReLU) + 1x1 mask conv (+bias).
// Per block: one (roi, m_tile, parity). 16 waves -> 16x256 deconv tile in
// LDS; 5 waves -> K=256, N=80 mask GEMM to f32 output.
// ---------------------------------------------------------------------------
__global__ __launch_bounds__(512) void deconv_mask_kernel(
    const bf16* __restrict__ xin, const bf16* __restrict__ wtp,
    const float* __restrict__ bt, const bf16* __restrict__ wmp,
    const float* __restrict__ bm, float* __restrict__ out, int nroi) {
  __shared__ bf16 zt[16 * 264];                 // 528B row stride (padded)

  const int lane = threadIdx.x & 31;
  const int wid  = threadIdx.x >> 5;            // 0..15
  long b = blockIdx.x;
  const int par    = (int)(b & 3); b >>= 2;     // ky*2+kx parity
  const int m_tile = (int)(b % 13);
  const int roi    = (int)(b / 13);
  if (roi >= nroi) return;

  const int row = lane & 15;
  const int hi  = lane >> 4;
  const int koffA = hi * 8;
  const int koffB = hi * 8;

  // ---- phase 1: z[m, wid*16..+15] = relu(y @ Wt_par + bt) -> LDS ----
  int m = m_tile * 16 + row;
  if (m > 195) m = 195;                         // clamp; rows discarded later
  const int oy = m / 14, ox = m % 14;
  const bf16* __restrict__ pa =
      xin + (long)roi * ROISTRIDE + ((oy + 1) * 16 + (ox + 1)) * 256 + koffA;
  const bf16* __restrict__ pb =
      wtp + ((long)par * 256 + wid * 16 + row) * 256 + koffB;

  v8f acc = {};
#pragma unroll
  for (int kk = 0; kk < 256; kk += 32) {
    V16 A, B;
    A.h[0] = *(const v8bf*)(pa + kk);
    A.h[1] = *(const v8bf*)(pa + kk + 16);
    B.h[0] = *(const v8bf*)(pb + kk);
    B.h[1] = *(const v8bf*)(pb + kk + 16);
    acc = __builtin_amdgcn_wmma_f32_16x16x32_bf16(
        false, A.v, false, B.v, (short)0, acc, false, false);
  }
  {
    const float bias = bt[wid * 16 + row];
#pragma unroll
    for (int r = 0; r < 8; ++r) {
      float v = acc[r] + bias;
      v = v > 0.f ? v : 0.f;
      zt[(r + hi * 8) * 264 + wid * 16 + row] = (bf16)v;
    }
  }
  __syncthreads();

  // ---- phase 2: out[m, :80] = z @ Wm + bm (5 waves, one 16x16 tile each) --
  if (wid < 5) {
    const bf16* __restrict__ wmrow = wmp + (long)(wid * 16 + row) * 256 + koffB;
    const bf16* __restrict__ za = &zt[row * 264 + koffA];
    v8f mac = {};
#pragma unroll
    for (int kk = 0; kk < 256; kk += 32) {
      V16 A, B;
      A.h[0] = *(const v8bf*)(za + kk);         // ds_load_b128
      A.h[1] = *(const v8bf*)(za + kk + 16);
      B.h[0] = *(const v8bf*)(wmrow + kk);
      B.h[1] = *(const v8bf*)(wmrow + kk + 16);
      mac = __builtin_amdgcn_wmma_f32_16x16x32_bf16(
          false, A.v, false, B.v, (short)0, mac, false, false);
    }
    const float bias = bm[wid * 16 + row];
    const int ky = par >> 1, kx = par & 1;
    int mr = m_tile * 16 + hi * 8;
    int ry = mr / 14, rx = mr - ry * 14;
#pragma unroll
    for (int r = 0; r < 8; ++r) {
      if (mr + r < 196) {
        const int opos = (2 * ry + ky) * 28 + (2 * rx + kx);
        out[((long)roi * 784 + opos) * 80 + wid * 16 + row] = mac[r] + bias;
      }
      ++rx;
      if (rx == 14) { rx = 0; ++ry; }
    }
  }
}

// ---------------------------------------------------------------------------
extern "C" void kernel_launch(void* const* d_in, const int* in_sizes, int n_in,
                              void* d_out, int out_size, void* d_ws, size_t ws_size,
                              hipStream_t stream) {
  const float* x  = (const float*)d_in[0];
  const float* Wc = (const float*)d_in[1];
  const float* bc = (const float*)d_in[2];
  const float* Wt = (const float*)d_in[3];
  const float* bt = (const float*)d_in[4];
  const float* Wm = (const float*)d_in[5];
  const float* bm = (const float*)d_in[6];
  float* out = (float*)d_out;

  const int NROI = in_sizes[0] / (14 * 14 * 256);   // 1024

  // workspace carve-up (all offsets 256B aligned)
  char* ws = (char*)d_ws;
  const size_t bufBytes = (size_t)NROI * ROISTRIDE * sizeof(bf16); // ~134 MB
  bf16* bufA = (bf16*)ws;
  bf16* bufB = (bf16*)(ws + bufBytes);
  bf16* wcp  = (bf16*)(ws + 2 * bufBytes);                         // 4*256*2304
  bf16* wtp  = (bf16*)(ws + 2 * bufBytes + 4718592);               // 4*256*256
  bf16* wmp  = (bf16*)(ws + 2 * bufBytes + 4718592 + 524288);      // 80*256

  // 1) zero ping-pong buffers (establish halos), convert + pack
  {
    long n16 = (long)(2 * bufBytes) / 16;
    zero_kernel<<<(int)((n16 + 255) / 256), 256, 0, stream>>>((uint4*)ws, n16);
    long n4 = (long)NROI * 196 * 64;
    cvt_pad_kernel<<<(int)((n4 + 255) / 256), 256, 0, stream>>>(x, bufA, NROI);
    pack_wc_kernel<<<(4 * 9 * 256 * 256 + 255) / 256, 256, 0, stream>>>(Wc, wcp);
    pack_wt_kernel<<<(4 * 256 * 256 + 255) / 256, 256, 0, stream>>>(Wt, wtp);
    pack_wm_kernel<<<(256 * 80 + 255) / 256, 256, 0, stream>>>(Wm, wmp);
  }

  // 2) four 3x3 conv layers, ping-pong bufA <-> bufB
  {
    long tiles = (long)NROI * 7 * 4;              // 32x64 tiles
    int blocks = (int)((tiles + 7) / 8);
    conv3x3_kernel<<<blocks, 256, 0, stream>>>(bufA, wcp + 0L * 256 * 2304, bc + 0 * 256, bufB, NROI);
    conv3x3_kernel<<<blocks, 256, 0, stream>>>(bufB, wcp + 1L * 256 * 2304, bc + 1 * 256, bufA, NROI);
    conv3x3_kernel<<<blocks, 256, 0, stream>>>(bufA, wcp + 2L * 256 * 2304, bc + 2 * 256, bufB, NROI);
    conv3x3_kernel<<<blocks, 256, 0, stream>>>(bufB, wcp + 3L * 256 * 2304, bc + 3 * 256, bufA, NROI);
  }

  // 3) fused transpose-conv + relu + 1x1 mask conv
  {
    int blocks = NROI * 13 * 4;
    deconv_mask_kernel<<<blocks, 512, 0, stream>>>(bufA, wtp, bt, wmp, bm, out, NROI);
  }
}
